// TransBottleneckSparse_39049842655656
// MI455X (gfx1250) — matrix operand order repaired and checked by reading, not compiled, not run on hardware
//
#include <hip/hip_runtime.h>

// ---------------------------------------------------------------------------
// TransBottleneckSparse for MI455X (gfx1250, wave32, WMMA bf16 16x16x32)
//
//   out1 = relu(x @ W1)                             [N,256]x[256,64]
//   out2 = relu(sum_k gather(out1,nbr_k) @ W2[k])   9 x [N,64]x[64,64]
//   out  = relu(out2 @ W3 + x)                      [N,64]x[64,256]
//
// v3 changes vs v2:
//  * all 4 N-tile B fragments are fetched from LDS before the WMMA burst so
//    the scheduler can use partial s_wait_dscnt and overlap ds_load of the
//    next fragment with the current WMMA pair (v2 drained DScnt per N-tile)
// ---------------------------------------------------------------------------

typedef __attribute__((ext_vector_type(16))) __bf16 v16bf;
typedef __attribute__((ext_vector_type(8)))  float  v8f;

union BF16Frag {
    uint4          u4[2];
    unsigned short h[16];
    __bf16         b[16];
    v16bf          v;
};

__device__ __forceinline__ unsigned short f32_to_bf16(float f) {
    __bf16 h = (__bf16)f;                        // lowers to v_cvt(_pk)_bf16_f32
    return __builtin_bit_cast(unsigned short, h);
}

__device__ __forceinline__ v8f wmma_bf16(const BF16Frag& a, const BF16Frag& b, v8f c) {
    return __builtin_amdgcn_wmma_f32_16x16x32_bf16(
        /*neg_a=*/false, a.v, /*neg_b=*/false, b.v,
        /*c_mod=*/(short)0, c, /*reuse_a=*/false, /*reuse_b=*/false);
}

// ---------------------------------------------------------------------------
// Pack a row-major f32 weight [K, NC] into B-fragment layout:
//   dst[((nt*KS + ks)*32 + lane)*16 + j] = bf16( W[ ks*32 + (j<8 ? h*8+j : 16+h*8+j-8),
//                                                   nt*16 + (lane&15) ] ), h = lane>>4
// (16-bit B 32x16 layout: lanes 0-15 K 0-7,16-23; lanes 16-31 K 8-15,24-31)
// ---------------------------------------------------------------------------
__global__ void k_pack_w(const float* __restrict__ W, unsigned short* __restrict__ dst,
                         int ncols, int ksteps, int total) {
    int i = blockIdx.x * blockDim.x + threadIdx.x;
    if (i >= total) return;
    int j    = i & 15;
    int lane = (i >> 4) & 31;
    int rest = i >> 9;
    int ks   = rest % ksteps;
    int nt   = rest / ksteps;
    int col  = nt * 16 + (lane & 15);
    int krow = ks * 32 + ((lane < 16) ? 0 : 8) + ((j < 8) ? j : 16 + (j - 8));
    dst[i] = f32_to_bf16(W[(size_t)krow * ncols + col]);
}

// cooperative global->LDS stage of packed weights (nu4 = # of uint4)
__device__ __forceinline__ void stage_lds(unsigned short* lds, const unsigned short* g, int nu4) {
    const uint4* src = (const uint4*)g;
    uint4*       dst = (uint4*)lds;
    for (int i = threadIdx.x; i < nu4; i += (int)blockDim.x) dst[i] = src[i];
    __syncthreads();
}

// ---------------------------------------------------------------------------
// conv1: out1 = relu(x @ W1).  8 waves/block, 2 M-tiles per wave.
// ---------------------------------------------------------------------------
__global__ void __launch_bounds__(256)
k_conv1(const float* __restrict__ x, const unsigned short* __restrict__ w1p,
        unsigned short* __restrict__ out1, int nrows) {
    __shared__ unsigned short w1s[16384];                 // 32 KB
    stage_lds(w1s, w1p, 2048);

    const int lane = threadIdx.x & 31;
    const int wave = threadIdx.x >> 5;
    const int pair = blockIdx.x * 8 + wave;               // covers 32 rows
    if (pair * 32 >= nrows) return;
    const int m0 = pair * 32;
    const int rl = lane & 15;
    const int ka = (lane < 16) ? 0 : 8;

    const float* xr0 = x + (size_t)(m0 + rl) * 256;
    const float* xr1 = x + (size_t)(m0 + 16 + rl) * 256;

    v8f acc[2][4] = {};

#pragma unroll
    for (int ks = 0; ks < 8; ++ks) {
        BF16Frag a0, a1;
#pragma unroll
        for (int q = 0; q < 2; ++q) {                     // two 8-half chunks
            const int ko = ks * 32 + ka + q * 16;
            float4 p0 = *(const float4*)(xr0 + ko);
            float4 p1 = *(const float4*)(xr0 + ko + 4);
            float4 q0 = *(const float4*)(xr1 + ko);
            float4 q1 = *(const float4*)(xr1 + ko + 4);
            a0.b[q * 8 + 0] = (__bf16)p0.x; a0.b[q * 8 + 1] = (__bf16)p0.y;
            a0.b[q * 8 + 2] = (__bf16)p0.z; a0.b[q * 8 + 3] = (__bf16)p0.w;
            a0.b[q * 8 + 4] = (__bf16)p1.x; a0.b[q * 8 + 5] = (__bf16)p1.y;
            a0.b[q * 8 + 6] = (__bf16)p1.z; a0.b[q * 8 + 7] = (__bf16)p1.w;
            a1.b[q * 8 + 0] = (__bf16)q0.x; a1.b[q * 8 + 1] = (__bf16)q0.y;
            a1.b[q * 8 + 2] = (__bf16)q0.z; a1.b[q * 8 + 3] = (__bf16)q0.w;
            a1.b[q * 8 + 4] = (__bf16)q1.x; a1.b[q * 8 + 5] = (__bf16)q1.y;
            a1.b[q * 8 + 6] = (__bf16)q1.z; a1.b[q * 8 + 7] = (__bf16)q1.w;
        }
        BF16Frag b[4];
#pragma unroll
        for (int nt = 0; nt < 4; ++nt) {                  // batch ds_loads
            const unsigned short* bp = w1s + ((((nt * 8) + ks) * 32 + lane) << 4);
            b[nt].u4[0] = *(const uint4*)(bp);
            b[nt].u4[1] = *(const uint4*)(bp + 8);
        }
#pragma unroll
        for (int nt = 0; nt < 4; ++nt) {                  // WMMA burst
            acc[0][nt] = wmma_bf16(a0, b[nt], acc[0][nt]);
            acc[1][nt] = wmma_bf16(a1, b[nt], acc[1][nt]);
        }
    }

    const int hi   = (lane >= 16) ? 8 : 0;
    const int colb = lane & 15;
#pragma unroll
    for (int t = 0; t < 2; ++t)
#pragma unroll
        for (int nt = 0; nt < 4; ++nt)
#pragma unroll
            for (int r = 0; r < 8; ++r) {
                float v = acc[t][nt][r];
                v = v > 0.0f ? v : 0.0f;
                out1[(size_t)(m0 + t * 16 + r + hi) * 64 + nt * 16 + colb] = f32_to_bf16(v);
            }
}

// ---------------------------------------------------------------------------
// conv2: out2 = relu(sum_k gather(out1, nbr[:,k]) @ W2[k]).
// 8 waves/block, 2 M-tiles per wave; A-row zeroed when neighbor < 0.
// ---------------------------------------------------------------------------
__global__ void __launch_bounds__(256)
k_conv2(const unsigned short* __restrict__ out1,
        const unsigned short* __restrict__ w2p,
        const int* __restrict__ nbr,
        unsigned short* __restrict__ out2, int nrows) {
    __shared__ unsigned short w2s[36864];                 // 72 KB
    stage_lds(w2s, w2p, 4608);

    const int lane = threadIdx.x & 31;
    const int wave = threadIdx.x >> 5;
    const int pair = blockIdx.x * 8 + wave;
    if (pair * 32 >= nrows) return;
    const int m0 = pair * 32;
    const int rl = lane & 15;
    const int ka = (lane < 16) ? 0 : 8;

    v8f acc[2][4] = {};

#pragma unroll
    for (int k = 0; k < 9; ++k) {
        const int n0 = nbr[(size_t)(m0 + rl) * 9 + k];
        const int n1 = nbr[(size_t)(m0 + 16 + rl) * 9 + k];
#pragma unroll
        for (int ks = 0; ks < 2; ++ks) {
            BF16Frag a0, a1;
            uint4 z; z.x = z.y = z.z = z.w = 0u;
            a0.u4[0] = z; a0.u4[1] = z;
            a1.u4[0] = z; a1.u4[1] = z;
            if (n0 >= 0) {
                const unsigned short* ap = out1 + (size_t)n0 * 64 + ks * 32 + ka;
                a0.u4[0] = *(const uint4*)(ap);
                a0.u4[1] = *(const uint4*)(ap + 16);
            }
            if (n1 >= 0) {
                const unsigned short* ap = out1 + (size_t)n1 * 64 + ks * 32 + ka;
                a1.u4[0] = *(const uint4*)(ap);
                a1.u4[1] = *(const uint4*)(ap + 16);
            }
            BF16Frag b[4];
#pragma unroll
            for (int nt = 0; nt < 4; ++nt) {              // batch ds_loads
                const unsigned short* bp =
                    w2s + (size_t)k * 4096 + ((((nt * 2) + ks) * 32 + lane) << 4);
                b[nt].u4[0] = *(const uint4*)(bp);
                b[nt].u4[1] = *(const uint4*)(bp + 8);
            }
#pragma unroll
            for (int nt = 0; nt < 4; ++nt) {              // WMMA burst
                acc[0][nt] = wmma_bf16(a0, b[nt], acc[0][nt]);
                acc[1][nt] = wmma_bf16(a1, b[nt], acc[1][nt]);
            }
        }
    }

    const int hi   = (lane >= 16) ? 8 : 0;
    const int colb = lane & 15;
#pragma unroll
    for (int t = 0; t < 2; ++t)
#pragma unroll
        for (int nt = 0; nt < 4; ++nt)
#pragma unroll
            for (int r = 0; r < 8; ++r) {
                float v = acc[t][nt][r];
                v = v > 0.0f ? v : 0.0f;
                out2[(size_t)(m0 + t * 16 + r + hi) * 64 + nt * 16 + colb] = f32_to_bf16(v);
            }
}

// ---------------------------------------------------------------------------
// conv3: out = relu(out2 @ W3 + x) -> f32 [N,256]
// 8 waves/block = 2 M-tiles; 4 waves share an M-tile, each covers 4 N-tiles.
// ---------------------------------------------------------------------------
__global__ void __launch_bounds__(256)
k_conv3(const unsigned short* __restrict__ out2,
        const unsigned short* __restrict__ w3p,
        const float* __restrict__ x,
        float* __restrict__ out, int nrows) {
    __shared__ unsigned short w3s[16384];                 // 32 KB
    stage_lds(w3s, w3p, 2048);

    const int lane   = threadIdx.x & 31;
    const int wave   = threadIdx.x >> 5;
    const int mtile  = blockIdx.x * 2 + (wave >> 2);
    const int ngroup = wave & 3;
    if (mtile * 16 >= nrows) return;
    const int mbase = mtile * 16;
    const int row   = mbase + (lane & 15);
    const int ka    = (lane < 16) ? 0 : 8;

    v8f acc[4] = {};

#pragma unroll
    for (int ks = 0; ks < 2; ++ks) {
        BF16Frag a;
        const unsigned short* ap = out2 + (size_t)row * 64 + ks * 32 + ka;
        a.u4[0] = *(const uint4*)(ap);
        a.u4[1] = *(const uint4*)(ap + 16);
        BF16Frag b[4];
#pragma unroll
        for (int t = 0; t < 4; ++t) {                     // batch ds_loads
            const int nt = ngroup * 4 + t;
            const unsigned short* bp = w3s + ((((nt * 2) + ks) * 32 + lane) << 4);
            b[t].u4[0] = *(const uint4*)(bp);
            b[t].u4[1] = *(const uint4*)(bp + 8);
        }
#pragma unroll
        for (int t = 0; t < 4; ++t)                       // WMMA burst
            acc[t] = wmma_bf16(a, b[t], acc[t]);
    }

    const int hi   = (lane >= 16) ? 8 : 0;
    const int colb = lane & 15;
#pragma unroll
    for (int t = 0; t < 4; ++t) {
        const int nt = ngroup * 4 + t;
#pragma unroll
        for (int r = 0; r < 8; ++r) {
            size_t oi = (size_t)(mbase + r + hi) * 256 + nt * 16 + colb;
            float v = acc[t][r] + x[oi];       // residual add
            out[oi] = v > 0.0f ? v : 0.0f;     // final relu
        }
    }
}

// ---------------------------------------------------------------------------
extern "C" void kernel_launch(void* const* d_in, const int* in_sizes, int n_in,
                              void* d_out, int out_size, void* d_ws, size_t ws_size,
                              hipStream_t stream) {
    const float* feat = (const float*)d_in[0];   // [N, 256]
    const float* W1   = (const float*)d_in[1];   // [256, 64]
    const float* W2   = (const float*)d_in[2];   // [9, 64, 64]
    const float* W3   = (const float*)d_in[3];   // [64, 256]
    const int*   nbr  = (const int*)  d_in[4];   // [N, 9]
    (void)n_in; (void)out_size; (void)ws_size;

    const int N     = in_sizes[0] / 256;         // 200000
    const int tiles = (N + 15) / 16;             // 12500
    const int pairs = (tiles + 1) / 2;           // 6250

    // workspace layout (bytes), 256B aligned
    char*  ws  = (char*)d_ws;
    size_t off = 0;
    auto alloc = [&](size_t bytes) { char* p = ws + off; off += (bytes + 255) & ~(size_t)255; return p; };
    unsigned short* out1b = (unsigned short*)alloc((size_t)N * 64 * 2);   // 25.6 MB
    unsigned short* out2b = (unsigned short*)alloc((size_t)N * 64 * 2);   // 25.6 MB
    unsigned short* w1p   = (unsigned short*)alloc(16384 * 2);            // 4nt*8ks*32*16
    unsigned short* w2p   = (unsigned short*)alloc(9 * 4096 * 2);         // 9*(4nt*2ks*32*16)
    unsigned short* w3p   = (unsigned short*)alloc(16384 * 2);            // 16nt*2ks*32*16

    // --- pack weights into B-fragment layout
    k_pack_w<<<(16384 + 255) / 256, 256, 0, stream>>>(W1, w1p, 64, 8, 16384);
    for (int k = 0; k < 9; ++k)
        k_pack_w<<<(4096 + 255) / 256, 256, 0, stream>>>(
            W2 + (size_t)k * 64 * 64, w2p + (size_t)k * 4096, 64, 2, 4096);
    k_pack_w<<<(16384 + 255) / 256, 256, 0, stream>>>(W3, w3p, 256, 2, 16384);

    // --- fused pipeline
    const int blk = 256;                                   // 8 wave32
    k_conv1<<<(pairs + 7) / 8, blk, 0, stream>>>(feat, w1p, out1b, N);
    k_conv2<<<(pairs + 7) / 8, blk, 0, stream>>>(out1b, w2p, nbr, out2b, N);
    k_conv3<<<(tiles + 1) / 2, blk, 0, stream>>>(out2b, w3p, feat, (float*)d_out, N);
}
